// LinearLoopLayer_21706764714263
// MI455X (gfx1250) — matrix operand order repaired and checked by reading, not compiled, not run on hardware
//
#include <hip/hip_runtime.h>
#include <stdint.h>

// ---------------------------------------------------------------------------
// Problem constants (all 4096, evenly divisible by every tile dim).
// ---------------------------------------------------------------------------
#define MAT     4096
#define TILE_M  128
#define TILE_N  128
#define TILE_K  32
#define NSTAGE  (MAT / TILE_K)   // 128 K-stages

typedef __attribute__((ext_vector_type(16))) __bf16 v16bf;
typedef __attribute__((ext_vector_type(8)))  float  v8f;

struct U128 { unsigned int u[4]; };            // 16-byte POD chunk
union  Frag { v16bf v; U128 q[2]; };           // 32-byte WMMA operand
struct US4  { unsigned short h[4]; };          // 8-byte bf16 quad

// Builtin's parameter type: pointer to 16-byte int vector, AS1 src / AS3 dst.
typedef int v4i __attribute__((vector_size(16)));
typedef __attribute__((address_space(1))) v4i g_v4i;   // global-AS 16B vector
typedef __attribute__((address_space(3))) v4i l_v4i;   // LDS-AS 16B vector

// ---------------------------------------------------------------------------
// CDNA5 async global->LDS copy (ASYNCcnt-tracked). Fallback: sync copy.
// ---------------------------------------------------------------------------
#if __has_builtin(__builtin_amdgcn_global_load_async_to_lds_b128)
#define HAVE_ASYNC 1
#define ASYNC_CP16(dst_lds, src_glb)                                         \
  __builtin_amdgcn_global_load_async_to_lds_b128(                            \
      (g_v4i*)(uintptr_t)(src_glb),                                          \
      (l_v4i*)(uint32_t)(uintptr_t)(dst_lds), 0, 0)
#else
#define HAVE_ASYNC 0
#define ASYNC_CP16(dst_lds, src_glb)                                         \
  (*(U128*)(dst_lds) = *(const U128*)(src_glb))
#endif

#if __has_builtin(__builtin_amdgcn_s_wait_asynccnt)
#define WAIT_ASYNC(n) __builtin_amdgcn_s_wait_asynccnt(n)
#else
#define WAIT_ASYNC(n) asm volatile("s_wait_asynccnt %0" :: "i"(n) : "memory")
#endif

// ---------------------------------------------------------------------------
// fp32 <-> bf16 bit helpers (round-to-nearest-even), no __bf16 arithmetic.
// ---------------------------------------------------------------------------
static __device__ __forceinline__ unsigned short f2bf_rne(float f) {
  unsigned u = __builtin_bit_cast(unsigned, f);
  u += 0x7FFFu + ((u >> 16) & 1u);
  return (unsigned short)(u >> 16);
}
static __device__ __forceinline__ float bf2f(unsigned short h) {
  unsigned u = ((unsigned)h) << 16;
  return __builtin_bit_cast(float, u);
}

// ---------------------------------------------------------------------------
// Kernel 1: split fp32 x / weight into bf16 (hi, lo) pairs. hi = rne(f),
// lo = rne(f - hi). hi+lo carries ~16 mantissa bits -> near-fp32 GEMM.
// Each thread handles 4 elements of x and 4 of weight (float4 in, 8B out).
// ---------------------------------------------------------------------------
__global__ __launch_bounds__(256) void split_bf16_kernel(
    const float* __restrict__ x, const float* __restrict__ w,
    unsigned short* __restrict__ Ahi, unsigned short* __restrict__ Alo,
    unsigned short* __restrict__ Whi, unsigned short* __restrict__ Wlo) {
  const size_t q    = (size_t)blockIdx.x * blockDim.x + threadIdx.x;
  const size_t base = q * 4;

  const float4 xv = *(const float4*)(x + base);
  const float4 wv = *(const float4*)(w + base);
  const float xs[4] = {xv.x, xv.y, xv.z, xv.w};
  const float ws4[4] = {wv.x, wv.y, wv.z, wv.w};

  US4 xh, xl, wh, wl;
#pragma unroll
  for (int i = 0; i < 4; ++i) {
    xh.h[i] = f2bf_rne(xs[i]);
    xl.h[i] = f2bf_rne(xs[i] - bf2f(xh.h[i]));
    wh.h[i] = f2bf_rne(ws4[i]);
    wl.h[i] = f2bf_rne(ws4[i] - bf2f(wh.h[i]));
  }
  *(US4*)(Ahi + base) = xh;
  *(US4*)(Alo + base) = xl;
  *(US4*)(Whi + base) = wh;
  *(US4*)(Wlo + base) = wl;
}

// ---------------------------------------------------------------------------
// Kernel 2: C = A * W^T + bias via V_WMMA_F32_16X16X32_BF16, bf16x3 scheme.
// Block = 128x128, 8 waves (4 m x 2 n), wave tile = 32x64 (2x4 WMMA tiles).
// Double-buffered 64 KB LDS filled by ASYNCcnt-tracked global->LDS copies.
// ---------------------------------------------------------------------------
__global__ __launch_bounds__(256) void wmma_gemm_bf16x3_kernel(
    const unsigned short* __restrict__ Ahi, const unsigned short* __restrict__ Alo,
    const unsigned short* __restrict__ Whi, const unsigned short* __restrict__ Wlo,
    const float* __restrict__ bias, float* __restrict__ out) {

  __shared__ unsigned short sAh[2][TILE_M][TILE_K];   // 16 KB
  __shared__ unsigned short sAl[2][TILE_M][TILE_K];   // 16 KB
  __shared__ unsigned short sWh[2][TILE_N][TILE_K];   // 16 KB
  __shared__ unsigned short sWl[2][TILE_N][TILE_K];   // 16 KB

  const int tid = threadIdx.x;
  const int m0  = blockIdx.y * TILE_M;   // batch rows
  const int n0  = blockIdx.x * TILE_N;   // out features

  // --- stage loader: 8 x 16B async copies per thread per stage -------------
  auto load_stage = [&](int kt, int buf) {
    const size_t k0 = (size_t)kt * TILE_K;
#pragma unroll
    for (int i = 0; i < 2; ++i) {
      const int c   = tid + 256 * i;        // 512 chunks of 8 bf16 per array
      const int row = c >> 2;               // 0..127
      const int col = (c & 3) * 8;          // 0,8,16,24
      const size_t ga = (size_t)(m0 + row) * MAT + k0 + col;
      const size_t gw = (size_t)(n0 + row) * MAT + k0 + col;
      ASYNC_CP16(&sAh[buf][row][col], Ahi + ga);
      ASYNC_CP16(&sAl[buf][row][col], Alo + ga);
      ASYNC_CP16(&sWh[buf][row][col], Whi + gw);
      ASYNC_CP16(&sWl[buf][row][col], Wlo + gw);
    }
  };

  // --- wave / lane geometry ------------------------------------------------
  const int lane = tid & 31;
  const int wave = tid >> 5;
  const int wm   = (wave & 3) * 32;     // wave row offset in block (4 waves)
  const int wn   = (wave >> 2) * 64;    // wave col offset in block (2 waves)
  const int half = lane >> 4;           // lane group 0 / 1
  const int l16  = lane & 15;

  v8f acc[2][4];
#pragma unroll
  for (int mt = 0; mt < 2; ++mt)
#pragma unroll
    for (int nt = 0; nt < 4; ++nt)
#pragma unroll
      for (int e = 0; e < 8; ++e) acc[mt][nt][e] = 0.0f;

  load_stage(0, 0);

  for (int kt = 0; kt < NSTAGE; ++kt) {
    const int buf = kt & 1;
    if (kt + 1 < NSTAGE) {
      load_stage(kt + 1, buf ^ 1);
      WAIT_ASYNC(8);    // async loads retire in order: stage kt is complete
    } else {
      WAIT_ASYNC(0);
    }
    __syncthreads();    // stage kt visible to all waves

    // --- LDS -> fragments (documented 16-bit WMMA lane/VGPR layouts) ------
    Frag aH[2], aL[2], bH[4], bL[4];
#pragma unroll
    for (int mt = 0; mt < 2; ++mt) {
      const int r  = wm + mt * 16 + l16;  // A row
      const int ka = half * 8;            // lanes 0-15: K 0-7 & 16-23; 16-31: 8-15 & 24-31
      aH[mt].q[0] = *(const U128*)&sAh[buf][r][ka];
      aH[mt].q[1] = *(const U128*)&sAh[buf][r][ka + 16];
      aL[mt].q[0] = *(const U128*)&sAl[buf][r][ka];
      aL[mt].q[1] = *(const U128*)&sAl[buf][r][ka + 16];
    }
#pragma unroll
    for (int nt = 0; nt < 4; ++nt) {
      const int r  = wn + nt * 16 + l16;  // B column = weight row
      const int kb = half * 8;            // lanes 0-15: K 0-7 & 16-23; 16-31: 8-15 & 24-31
      bH[nt].q[0] = *(const U128*)&sWh[buf][r][kb];
      bH[nt].q[1] = *(const U128*)&sWh[buf][r][kb + 16];
      bL[nt].q[0] = *(const U128*)&sWl[buf][r][kb];
      bL[nt].q[1] = *(const U128*)&sWl[buf][r][kb + 16];
    }

    // --- bf16x3: acc += AhWh + AhWl + AlWh (fp32 accumulate) --------------
#pragma unroll
    for (int mt = 0; mt < 2; ++mt)
#pragma unroll
      for (int nt = 0; nt < 4; ++nt) {
        acc[mt][nt] = __builtin_amdgcn_wmma_f32_16x16x32_bf16(
            false, aH[mt].v, false, bH[nt].v, (short)0, acc[mt][nt], false, false);
        acc[mt][nt] = __builtin_amdgcn_wmma_f32_16x16x32_bf16(
            false, aH[mt].v, false, bL[nt].v, (short)0, acc[mt][nt], false, false);
        acc[mt][nt] = __builtin_amdgcn_wmma_f32_16x16x32_bf16(
            false, aL[mt].v, false, bH[nt].v, (short)0, acc[mt][nt], false, false);
      }

    __syncthreads();    // everyone done reading buf before it is re-filled
  }

  // --- epilogue: bias add + fp32 store ------------------------------------
  // C/D layout: VGPR r holds M=r (lanes 0-15) / M=r+8 (lanes 16-31), N=l16.
#pragma unroll
  for (int mt = 0; mt < 2; ++mt)
#pragma unroll
    for (int nt = 0; nt < 4; ++nt) {
      const int col   = n0 + wn + nt * 16 + l16;
      const float bv  = bias[col];
      const int  row0 = m0 + wm + mt * 16 + half * 8;
#pragma unroll
      for (int r = 0; r < 8; ++r)
        out[(size_t)(row0 + r) * MAT + col] = acc[mt][nt][r] + bv;
    }
}

// ---------------------------------------------------------------------------
// Host-side launcher. Inputs: x [4096,4096] f32, weight [4096,4096] f32,
// bias [4096] f32. Output: [4096,4096] f32. Uses 128 MB of d_ws for the
// bf16 hi/lo operand planes (written fresh every call -> deterministic).
// ---------------------------------------------------------------------------
extern "C" void kernel_launch(void* const* d_in, const int* in_sizes, int n_in,
                              void* d_out, int out_size, void* d_ws, size_t ws_size,
                              hipStream_t stream) {
  (void)in_sizes; (void)n_in; (void)out_size; (void)ws_size;

  const float* x    = (const float*)d_in[0];
  const float* w    = (const float*)d_in[1];
  const float* bias = (const float*)d_in[2];
  float*       out  = (float*)d_out;

  const size_t EM = (size_t)MAT * (size_t)MAT;
  unsigned short* Ahi = (unsigned short*)d_ws;
  unsigned short* Alo = Ahi + EM;
  unsigned short* Whi = Alo + EM;
  unsigned short* Wlo = Whi + EM;

  // 1) fp32 -> bf16 hi/lo split (memory-bound pre-pass, ~11 us at HBM peak)
  const int split_blocks = (int)(EM / 4 / 256);   // 16384
  split_bf16_kernel<<<split_blocks, 256, 0, stream>>>(x, w, Ahi, Alo, Whi, Wlo);

  // 2) WMMA GEMM, 32x32 grid of 128x128 tiles
  dim3 grid(MAT / TILE_N, MAT / TILE_M);
  wmma_gemm_bf16x3_kernel<<<grid, 256, 0, stream>>>(Ahi, Alo, Whi, Wlo, bias, out);
}
